// EdgeFluxModule_23974507446372
// MI455X (gfx1250) — compile-verified, matrix operand order checked by reading, not run on hardware
//
#include <hip/hip_runtime.h>
#include <hip/hip_bf16.h>
#include <math.h>

// ---------------------------------------------------------------------------
// CDNA5 (gfx1250) wave32 WMMA implementation of the EdgeFlux GNN.
//   prep_weights : swizzle all weight matrices into B-fragment layout (d_ws)
//   node_kernel  : h = phi_node(x); g = phi1(h)+phi2(h)   (WMMA, 16 nodes/wave)
//   edge_kernel  : v = g[src]+g[dst]; eps; m = phi_msg([v,eps]); fused psi heads
//                  + geometry outputs                      (WMMA, 16 edges/wave)
// ---------------------------------------------------------------------------

typedef __attribute__((ext_vector_type(16))) _Float16 v16h;
typedef __attribute__((ext_vector_type(8)))  _Float16 v8h;
typedef __attribute__((ext_vector_type(4)))  _Float16 v4h;
typedef __attribute__((ext_vector_type(8)))  float    v8f;

// Weight-fragment regions (offsets in halves inside d_ws).
// Each fragment = 32 lanes x 16 halves = 512 halves (one 16x16x32 B operand).
#define OFF_EDGE2 0        //  32x32  phi_edge L2      (1x2  frags)
#define OFF_MSG1  1024     //  96x128 phi_msg  L1      (3x8)
#define OFF_MSG2  13312    // 128x64  phi_msg  L2      (4x4)
#define OFF_H1    21504    //  64x256 fused psi hidden (2x16)
#define OFF_H2    37888    // 256x16  block-diag psi out (8x1)
#define OFF_NODE1 41984    //  32x64  phi_node L1 (K padded 5->32) (1x4)
#define OFF_NODE2 44032    //  64x64  phi_node L2      (2x4)
#define OFF_P1A   48128    //  64x32  phi1 L1          (2x2)
#define OFF_P1B   50176    //  32x64  phi1 L2          (1x4)
#define OFF_P2A   52224    //  64x32  phi2 L1          (2x2)
#define OFF_P2B   54272    //  32x64  phi2 L2          (1x4)
#define WF_HALVES 56320
#define BIASH1_BYTES_OFF 114688   // float[256]: concat psi hidden biases
#define G_BYTES_OFF      131072   // _Float16[N*64]: per-node g = phi1(h)+phi2(h)

#define WMMA_F16(a, b, c) \
  __builtin_amdgcn_wmma_f32_16x16x32_f16(false, (a), false, (b), (short)0, (c), false, false)

// Branch-free GELU (tanh form, algebraically reduced to a sigmoid):
//   gelu(x) ~= x * sigmoid(1.59576912*x + 0.07135481*x^3)
// 9 straight-line VALU ops (v_exp_f32 + v_rcp_f32 transcendentals), no EXEC
// branching -- replaces the ~45-instruction branchy libm erff expansion that
// dominated the VALU histogram.
__device__ __forceinline__ float gelu_fast(float x) {
  float x3 = x * x * x;
  float u  = 0.7978845608f * x + 0.0356774081f * x3;   // sqrt(2/pi)*(x+0.044715x^3)
  float t  = u * -2.885390082f;                        // -2u * log2(e)
  float e  = __builtin_amdgcn_exp2f(t);                // exp(-2u)
  float s  = __builtin_amdgcn_rcpf(1.0f + e);          // sigmoid(2u)
  return x * s;
}

// A-fragment (16x32 f16, M=edges/nodes, K=features) from a row-major f16 LDS
// buffer: lane<16 holds k 0..7 and 16..23 of row m=lane; lanes>=16 hold
// k 8..15 and 24..31 (ISA 7.12.2). Two contiguous b128 LDS loads per lane.
__device__ __forceinline__ v16h load_afrag(const _Float16* buf, int ld, int lane) {
  int m = lane & 15;
  int kbase = (lane < 16) ? 0 : 8;
  const _Float16* row = buf + m * ld;
  v8h lo = *(const v8h*)(row + kbase);        // k = kbase .. kbase+7
  v8h hi = *(const v8h*)(row + kbase + 16);   // k = kbase+16 .. kbase+23
  return __builtin_shufflevector(lo, hi, 0, 1, 2, 3, 4, 5, 6, 7,
                                 8, 9, 10, 11, 12, 13, 14, 15);
}

// B-fragment from the pre-swizzled global fragment table (2x16B loads/lane).
__device__ __forceinline__ v16h load_bfrag(const _Float16* region, int fidx, int lane) {
  const v8h* p = (const v8h*)(region + (size_t)fidx * 512 + lane * 16);
  v8h lo = p[0], hi = p[1];
  return __builtin_shufflevector(lo, hi, 0, 1, 2, 3, 4, 5, 6, 7,
                                 8, 9, 10, 11, 12, 13, 14, 15);
}

// D tile (16x16 f32; lane<16 holds col n=lane rows v.., lanes>=16 rows v+8)
// -> bias (+optional GELU) -> f16 row-major LDS buffer at column c0.
__device__ __forceinline__ void store_tile(v8f acc, _Float16* buf, int ld, int c0,
                                           int lane, const float* bias, bool do_gelu) {
  int n = lane & 15;
  int mb = (lane < 16) ? 0 : 8;
  float b = bias[c0 + n];
#pragma unroll
  for (int v = 0; v < 8; ++v) {
    float x = acc[v] + b;
    if (do_gelu) x = gelu_fast(x);
    buf[(mb + v) * ld + c0 + n] = (_Float16)x;
  }
}

// Scatter one weight element W[k][n] into its B-fragment slot.
__device__ __forceinline__ void put_frag(_Float16* base, int ntiles, int k, int n, float v) {
  int kt = k >> 5, kk = k & 31, ntl = n >> 4, nn = n & 15;
  int lane = nn + ((kk >> 3) & 1) * 16;
  int t = (kk & 7) + ((kk & 16) ? 8 : 0);
  base[((kt * ntiles + ntl) * 32 + lane) * 16 + t] = (_Float16)v;
}

__device__ void stage_mat(const float* W, int K, int Ncols, _Float16* dst, int tid, int nth) {
  int ntiles = Ncols >> 4;
  for (int idx = tid; idx < K * Ncols; idx += nth) {
    int k = idx / Ncols, n = idx % Ncols;
    put_frag(dst, ntiles, k, n, W[idx]);
  }
}

// ---------------------------------------------------------------------------
__global__ void prep_weights(_Float16* __restrict__ wf, float* __restrict__ biasH1,
    const float* ew1, const float* mw0, const float* mw1,
    const float* rw0, const float* qw0, const float* pw0, const float* uw0,
    const float* rw1, const float* qw1, const float* pw1, const float* uw1,
    const float* nw0, const float* nw1,
    const float* p1w0, const float* p1w1, const float* p2w0, const float* p2w1,
    const float* rb0, const float* qb0, const float* pb0, const float* ub0) {
  int tid = blockIdx.x * blockDim.x + threadIdx.x;
  int nth = gridDim.x * blockDim.x;
  stage_mat(ew1, 32, 32, wf + OFF_EDGE2, tid, nth);
  stage_mat(mw0, 96, 128, wf + OFF_MSG1, tid, nth);
  stage_mat(mw1, 128, 64, wf + OFF_MSG2, tid, nth);
  // fused psi hidden: concat 4x (64x64) along N -> 64x256
  for (int idx = tid; idx < 64 * 256; idx += nth) {
    int k = idx / 256, n = idx % 256;
    const float* src = (n < 64) ? rw0 : (n < 128) ? qw0 : (n < 192) ? pw0 : uw0;
    put_frag(wf + OFF_H1, 16, k, n, src[k * 64 + (n & 63)]);
  }
  // fused psi output: block-diagonal 256x16 (cols 0..4 used)
  for (int idx = tid; idx < 256 * 16; idx += nth) {
    int k = idx / 16, n = idx % 16;
    int blk = k >> 6, kk = k & 63;
    float v = 0.f;
    if (blk == 0 && n == 0) v = rw1[kk];
    else if (blk == 1 && n == 1) v = qw1[kk];
    else if (blk == 2 && n == 2) v = pw1[kk];
    else if (blk == 3 && (n == 3 || n == 4)) v = uw1[kk * 2 + (n - 3)];
    put_frag(wf + OFF_H2, 1, k, n, v);
  }
  // phi_node L1: K padded 5 -> 32 with zeros
  for (int idx = tid; idx < 32 * 64; idx += nth) {
    int k = idx / 64, n = idx % 64;
    put_frag(wf + OFF_NODE1, 4, k, n, (k < 5) ? nw0[k * 64 + n] : 0.f);
  }
  stage_mat(nw1, 64, 64, wf + OFF_NODE2, tid, nth);
  stage_mat(p1w0, 64, 32, wf + OFF_P1A, tid, nth);
  stage_mat(p1w1, 32, 64, wf + OFF_P1B, tid, nth);
  stage_mat(p2w0, 64, 32, wf + OFF_P2A, tid, nth);
  stage_mat(p2w1, 32, 64, wf + OFF_P2B, tid, nth);
  for (int n = tid; n < 256; n += nth)
    biasH1[n] = (n < 64) ? rb0[n] : (n < 128) ? qb0[n - 64]
              : (n < 192) ? pb0[n - 128] : ub0[n - 192];
}

// ---------------------------------------------------------------------------
// Node pass: 16 nodes per wave. g[node][64] (f16) = phi1(h)+phi2(h).
__global__ __launch_bounds__(128) void node_kernel(
    const float* __restrict__ xu, const _Float16* __restrict__ wf,
    const float* __restrict__ nb0, const float* __restrict__ nb1,
    const float* __restrict__ p1b0, const float* __restrict__ p1b1,
    const float* __restrict__ p2b0, const float* __restrict__ p2b1,
    _Float16* __restrict__ g, int N) {
  __shared__ __align__(16) _Float16 sA[4][16 * 64];
  __shared__ __align__(16) _Float16 sB[4][16 * 64];
  const int wave = threadIdx.x >> 5, lane = threadIdx.x & 31;
  const int n0 = (blockIdx.x * 4 + wave) * 16;
  if (n0 >= N) return;
  _Float16* A  = sA[wave];
  _Float16* Bq = sB[wave];
  {  // stage node features, zero-padded K: 5 -> 32
    int m = lane & 15, node = n0 + m;
    int jb = (lane >> 4) * 16;
#pragma unroll
    for (int jj = 0; jj < 16; ++jj) {
      int j = jb + jj;
      float v = (j < 5 && node < N) ? xu[node * 5 + j] : 0.f;
      A[m * 64 + j] = (_Float16)v;
    }
  }
  __builtin_amdgcn_wave_barrier();
  {  // phi_node L1 (pad32 -> 64, gelu): A[:, :32] -> Bq
    v16h a0 = load_afrag(A, 64, lane);
#pragma unroll
    for (int nt = 0; nt < 4; ++nt) {
      v8f acc = {};
      acc = WMMA_F16(a0, load_bfrag(wf + OFF_NODE1, nt, lane), acc);
      store_tile(acc, Bq, 64, nt * 16, lane, nb0, true);
    }
  }
  __builtin_amdgcn_wave_barrier();
  {  // phi_node L2 (64 -> 64): Bq -> A  (h)
    v16h h0 = load_afrag(Bq, 64, lane);
    v16h h1 = load_afrag(Bq + 32, 64, lane);
#pragma unroll
    for (int nt = 0; nt < 4; ++nt) {
      v8f acc = {};
      acc = WMMA_F16(h0, load_bfrag(wf + OFF_NODE2, 0 * 4 + nt, lane), acc);
      acc = WMMA_F16(h1, load_bfrag(wf + OFF_NODE2, 1 * 4 + nt, lane), acc);
      store_tile(acc, A, 64, nt * 16, lane, nb1, false);
    }
  }
  __builtin_amdgcn_wave_barrier();
  v16h ha = load_afrag(A, 64, lane);
  v16h hb = load_afrag(A + 32, 64, lane);
  {  // phi1 L1 (64 -> 32, gelu) -> Bq[:, :32]
#pragma unroll
    for (int nt = 0; nt < 2; ++nt) {
      v8f acc = {};
      acc = WMMA_F16(ha, load_bfrag(wf + OFF_P1A, 0 * 2 + nt, lane), acc);
      acc = WMMA_F16(hb, load_bfrag(wf + OFF_P1A, 1 * 2 + nt, lane), acc);
      store_tile(acc, Bq, 64, nt * 16, lane, p1b0, true);
    }
  }
  __builtin_amdgcn_wave_barrier();
  v8f gacc[4];
  {  // phi1 L2 (32 -> 64) into registers
    v16h t1 = load_afrag(Bq, 64, lane);
#pragma unroll
    for (int nt = 0; nt < 4; ++nt) {
      v8f z = {};
      gacc[nt] = WMMA_F16(t1, load_bfrag(wf + OFF_P1B, nt, lane), z);
    }
  }
  {  // phi2 L1 (64 -> 32, gelu) -> Bq[:, 32:64]
#pragma unroll
    for (int nt = 0; nt < 2; ++nt) {
      v8f acc = {};
      acc = WMMA_F16(ha, load_bfrag(wf + OFF_P2A, 0 * 2 + nt, lane), acc);
      acc = WMMA_F16(hb, load_bfrag(wf + OFF_P2A, 1 * 2 + nt, lane), acc);
      store_tile(acc, Bq, 64, 32 + nt * 16, lane, p2b0, true);
    }
  }
  __builtin_amdgcn_wave_barrier();
  {  // phi2 L2 accumulated on top of phi1 L2
    v16h t2 = load_afrag(Bq + 32, 64, lane);
#pragma unroll
    for (int nt = 0; nt < 4; ++nt)
      gacc[nt] = WMMA_F16(t2, load_bfrag(wf + OFF_P2B, nt, lane), gacc[nt]);
  }
  {  // g = gacc + (b1_phi1 + b1_phi2), stored f16
    int n = lane & 15, mb = (lane < 16) ? 0 : 8;
#pragma unroll
    for (int nt = 0; nt < 4; ++nt) {
      int col = nt * 16 + n;
      float bias = p1b1[col] + p2b1[col];
#pragma unroll
      for (int v = 0; v < 8; ++v) {
        int node = n0 + mb + v;
        if (node < N) g[(long long)node * 64 + col] = (_Float16)(gacc[nt][v] + bias);
      }
    }
  }
}

// ---------------------------------------------------------------------------
// Edge pass: 16 edges per wave, 82 WMMAs per tile.
__global__ __launch_bounds__(128) void edge_kernel(
    const _Float16* __restrict__ g, const long long* __restrict__ ei,
    const float* __restrict__ eattr, const _Float16* __restrict__ wf,
    const float* __restrict__ biasH1,
    const float* __restrict__ ew0, const float* __restrict__ eb0,
    const float* __restrict__ eb1, const float* __restrict__ mb0,
    const float* __restrict__ mb1,
    const float* __restrict__ rb1, const float* __restrict__ qb1,
    const float* __restrict__ pb1, const float* __restrict__ ub1,
    float* __restrict__ out, long long E) {
  __shared__ __align__(16) _Float16 sA[4][16 * 128];
  __shared__ __align__(16) _Float16 sB[4][16 * 256];
  __shared__ __align__(16) float    sO[4][16 * 16];
  __shared__ int sIdx[4][32];   // [0:16) = src, [16:32) = dst node ids
  const int wave = threadIdx.x >> 5, lane = threadIdx.x & 31;
  const long long e0 = ((long long)blockIdx.x * 4 + wave) * 16;
  if (e0 >= E) return;
  _Float16* A  = sA[wave];
  _Float16* Bx = sB[wave];
  float*    O  = sO[wave];

  // Warm the 112KB fragment table into WGP$/L2 (global_prefetch_b8).
  for (int i = threadIdx.x; i < (WF_HALVES * 2) / 128; i += 128)
    __builtin_prefetch(((const char*)wf) + i * 128, 0, 3);

  // Stage the 32 unique edge endpoints once (lanes 0-15: src, 16-31: dst).
  {
    long long e = e0 + (lane & 15);
    long long off = (lane < 16) ? 0 : E;
    sIdx[wave][lane] = (e < E) ? (int)ei[off + e] : 0;
  }
  __builtin_amdgcn_wave_barrier();

  // v = g[src] + g[dst]  (packed f16 adds) -> A[:, 0:64)
#pragma unroll
  for (int r2 = 0; r2 < 8; ++r2) {
    int m = r2 * 2 + (lane >> 4);
    int c = (lane & 15) * 4;
    long long e = e0 + m;
    v4h val = {};
    if (e < E) {
      long long s = sIdx[wave][m];
      long long d = sIdx[wave][16 + m];
      v4h a = *(const v4h*)(g + s * 64 + c);
      v4h b = *(const v4h*)(g + d * 64 + c);
      val = a + b;
    }
    *(v4h*)(A + m * 128 + c) = val;
  }
  {  // phi_edge L1 (1 -> 32, gelu): VALU, -> A[:, 96:128)
    int m = lane & 15;
    long long e = e0 + m;
    float rv = (e < E) ? eattr[e * 3 + 2] : 0.f;
    int jb = (lane >> 4) * 16;
#pragma unroll
    for (int jj = 0; jj < 16; ++jj) {
      int j = jb + jj;
      A[m * 128 + 96 + j] = (_Float16)gelu_fast(ew0[j] * rv + eb0[j]);
    }
  }
  __builtin_amdgcn_wave_barrier();
  {  // phi_edge L2 (32 -> 32): eps -> A[:, 64:96)
    v16h at = load_afrag(A + 96, 128, lane);
#pragma unroll
    for (int nt = 0; nt < 2; ++nt) {
      v8f acc = {};
      acc = WMMA_F16(at, load_bfrag(wf + OFF_EDGE2, nt, lane), acc);
      store_tile(acc, A, 128, 64 + nt * 16, lane, eb1, false);
    }
  }
  __builtin_amdgcn_wave_barrier();
  {  // phi_msg L1 (96 -> 128, gelu): [v|eps] -> A[:, 0:128)
    v16h a0 = load_afrag(A, 128, lane);
    v16h a1 = load_afrag(A + 32, 128, lane);
    v16h a2 = load_afrag(A + 64, 128, lane);
#pragma unroll
    for (int nt = 0; nt < 8; ++nt) {
      v8f acc = {};
      acc = WMMA_F16(a0, load_bfrag(wf + OFF_MSG1, 0 * 8 + nt, lane), acc);
      acc = WMMA_F16(a1, load_bfrag(wf + OFF_MSG1, 1 * 8 + nt, lane), acc);
      acc = WMMA_F16(a2, load_bfrag(wf + OFF_MSG1, 2 * 8 + nt, lane), acc);
      store_tile(acc, A, 128, nt * 16, lane, mb0, true);
    }
  }
  __builtin_amdgcn_wave_barrier();
  {  // phi_msg L2 (128 -> 64): m -> A[:, 0:64)
    v16h h0 = load_afrag(A, 128, lane);
    v16h h1 = load_afrag(A + 32, 128, lane);
    v16h h2 = load_afrag(A + 64, 128, lane);
    v16h h3 = load_afrag(A + 96, 128, lane);
#pragma unroll
    for (int nt = 0; nt < 4; ++nt) {
      v8f acc = {};
      acc = WMMA_F16(h0, load_bfrag(wf + OFF_MSG2, 0 * 4 + nt, lane), acc);
      acc = WMMA_F16(h1, load_bfrag(wf + OFF_MSG2, 1 * 4 + nt, lane), acc);
      acc = WMMA_F16(h2, load_bfrag(wf + OFF_MSG2, 2 * 4 + nt, lane), acc);
      acc = WMMA_F16(h3, load_bfrag(wf + OFF_MSG2, 3 * 4 + nt, lane), acc);
      store_tile(acc, A, 128, nt * 16, lane, mb1, false);
    }
  }
  __builtin_amdgcn_wave_barrier();
  {  // fused psi hidden (64 -> 256, gelu): -> Bx[:, 0:256)
    v16h m0 = load_afrag(A, 128, lane);
    v16h m1 = load_afrag(A + 32, 128, lane);
#pragma unroll
    for (int nt = 0; nt < 16; ++nt) {
      v8f acc = {};
      acc = WMMA_F16(m0, load_bfrag(wf + OFF_H1, 0 * 16 + nt, lane), acc);
      acc = WMMA_F16(m1, load_bfrag(wf + OFF_H1, 1 * 16 + nt, lane), acc);
      store_tile(acc, Bx, 256, nt * 16, lane, biasH1, true);
    }
  }
  __builtin_amdgcn_wave_barrier();
  {  // fused psi output (block-diagonal 256 -> 16; cols 0..4 useful)
    v8f acc = {};
#pragma unroll
    for (int kt = 0; kt < 8; ++kt) {
      v16h a = load_afrag(Bx + kt * 32, 256, lane);
      acc = WMMA_F16(a, load_bfrag(wf + OFF_H2, kt, lane), acc);
    }
    int n = lane & 15, mb = (lane < 16) ? 0 : 8;
#pragma unroll
    for (int v = 0; v < 8; ++v) O[(mb + v) * 16 + n] = acc[v];
  }
  __builtin_amdgcn_wave_barrier();
  // Geometry + final stores (one edge per lane, lanes 0..15).
  if (lane < 16) {
    long long e = e0 + lane;
    if (e < E) {
      float dx = eattr[e * 3 + 0], dy = eattr[e * 3 + 1], rv = eattr[e * 3 + 2];
      float inv = 1.f / (rv + 1e-12f);
      float nx = dx * inv, ny = dy * inv;
      float tx = -ny, ty = nx;
      float a_rho = O[lane * 16 + 0] + rb1[0];
      float a_e   = O[lane * 16 + 1] + qb1[0];
      float a_p   = O[lane * 16 + 2] + pb1[0];
      float u0    = O[lane * 16 + 3] + ub1[0];
      float u1    = O[lane * 16 + 4] + ub1[1];
      size_t Es = (size_t)E;
      out[e] = a_rho;
      out[Es + e] = a_e;
      out[2 * Es + e] = a_p;
      out[3 * Es + 2 * e + 0] = a_rho * nx;  out[3 * Es + 2 * e + 1] = a_rho * ny;
      out[5 * Es + 2 * e + 0] = a_e * nx;    out[5 * Es + 2 * e + 1] = a_e * ny;
      out[7 * Es + 2 * e + 0] = a_p * nx;    out[7 * Es + 2 * e + 1] = a_p * ny;
      out[9 * Es + 2 * e + 0] = u0 * nx + u1 * tx;
      out[9 * Es + 2 * e + 1] = u0 * ny + u1 * ty;
      out[11 * Es + 2 * e + 0] = nx;         out[11 * Es + 2 * e + 1] = ny;
      out[13 * Es + e] = rv;
    }
  }
}

// ---------------------------------------------------------------------------
extern "C" void kernel_launch(void* const* d_in, const int* in_sizes, int n_in,
                              void* d_out, int out_size, void* d_ws, size_t ws_size,
                              hipStream_t stream) {
  const float* node_u   = (const float*)d_in[0];
  const long long* ei   = (const long long*)d_in[1];  // int64 edge_index
  const float* eattr    = (const float*)d_in[2];
  // params in setup_inputs dict order: (w0,b0,w1,b1) per MLP
  const float* nw0 = (const float*)d_in[3];  const float* nb0 = (const float*)d_in[4];
  const float* nw1 = (const float*)d_in[5];  const float* nb1 = (const float*)d_in[6];
  const float* p1w0 = (const float*)d_in[7];  const float* p1b0 = (const float*)d_in[8];
  const float* p1w1 = (const float*)d_in[9];  const float* p1b1 = (const float*)d_in[10];
  const float* p2w0 = (const float*)d_in[11]; const float* p2b0 = (const float*)d_in[12];
  const float* p2w1 = (const float*)d_in[13]; const float* p2b1 = (const float*)d_in[14];
  const float* ew0 = (const float*)d_in[15]; const float* eb0 = (const float*)d_in[16];
  const float* ew1 = (const float*)d_in[17]; const float* eb1 = (const float*)d_in[18];
  const float* mw0 = (const float*)d_in[19]; const float* mb0 = (const float*)d_in[20];
  const float* mw1 = (const float*)d_in[21]; const float* mb1 = (const float*)d_in[22];
  const float* rw0 = (const float*)d_in[23]; const float* rb0 = (const float*)d_in[24];
  const float* rw1 = (const float*)d_in[25]; const float* rb1 = (const float*)d_in[26];
  const float* qw0 = (const float*)d_in[27]; const float* qb0 = (const float*)d_in[28];
  const float* qw1 = (const float*)d_in[29]; const float* qb1 = (const float*)d_in[30];
  const float* pw0 = (const float*)d_in[31]; const float* pb0 = (const float*)d_in[32];
  const float* pw1 = (const float*)d_in[33]; const float* pb1 = (const float*)d_in[34];
  const float* uw0 = (const float*)d_in[35]; const float* ub0 = (const float*)d_in[36];
  const float* uw1 = (const float*)d_in[37]; const float* ub1 = (const float*)d_in[38];

  int N = in_sizes[0] / 5;
  long long E = in_sizes[1] / 2;

  char* ws = (char*)d_ws;
  _Float16* wf = (_Float16*)ws;
  float* biasH1 = (float*)(ws + BIASH1_BYTES_OFF);
  _Float16* g = (_Float16*)(ws + G_BYTES_OFF);   // needs ~128KB + N*128 bytes

  prep_weights<<<64, 256, 0, stream>>>(wf, biasH1,
      ew1, mw0, mw1, rw0, qw0, pw0, uw0, rw1, qw1, pw1, uw1,
      nw0, nw1, p1w0, p1w1, p2w0, p2w1, rb0, qb0, pb0, ub0);

  int ntiles = (N + 15) / 16;
  node_kernel<<<(ntiles + 3) / 4, 128, 0, stream>>>(
      node_u, wf, nb0, nb1, p1b0, p1b1, p2b0, p2b1, g, N);

  long long etiles = (E + 15) / 16;
  edge_kernel<<<(int)((etiles + 3) / 4), 128, 0, stream>>>(
      g, ei, eattr, wf, biasH1, ew0, eb0, eb1, mb0, mb1,
      rb1, qb1, pb1, ub1, (float*)d_out, E);
}
